// Processor_11708080848934
// MI455X (gfx1250) — matrix operand order repaired and checked by reading, not compiled, not run on hardware
//
#include <hip/hip_runtime.h>

// ---------------------------------------------------------------------------
// Problem constants (match the JAX reference)
// ---------------------------------------------------------------------------
#define NN 40000
#define EE 640000
#define DD 128
#define MM 10
#define LN_EPS 1e-5f

#define RT 32           // rows (edges / nodes) per block: 2 WMMA row-tiles
#define KT 32           // K per WMMA step (bf16 16x16x32)
#define NTHREADS 256    // 8 waves: one 16-col WMMA tile per wave
#define TILE_HALVES (DD * KT)   // one staged B tile: 4096 bf16 = 8 KB

typedef __bf16        v16bf __attribute__((ext_vector_type(16)));
typedef float         v8f   __attribute__((ext_vector_type(8)));
typedef unsigned int  v4u   __attribute__((ext_vector_type(4)));

// exact pointee type of the async-DMA builtin params (per hipcc diagnostic)
typedef int vi4 __attribute__((vector_size(4 * sizeof(int))));
typedef __attribute__((address_space(1))) vi4 gvi4;   // global (AS1 / __device__)
typedef __attribute__((address_space(3))) vi4 lvi4;   // LDS (AS3 / __shared__)

#if __has_builtin(__builtin_amdgcn_global_load_async_to_lds_b128) && \
    __has_builtin(__builtin_amdgcn_s_wait_asynccnt)
#define USE_ASYNC 1
#else
#define USE_ASYNC 0
#endif

// fp32 -> bf16 round-to-nearest-even
__device__ __forceinline__ unsigned short f2bf(float f) {
    unsigned u = __builtin_bit_cast(unsigned, f);
    u += 0x7FFFu + ((u >> 16) & 1u);
    return (unsigned short)(u >> 16);
}
__device__ __forceinline__ unsigned pack2bf(float a, float b) {
    return (unsigned)f2bf(a) | ((unsigned)f2bf(b) << 16);
}

// two 16-byte LDS loads -> one 16-element bf16 fragment
__device__ __forceinline__ v16bf load_frag_pair(const unsigned short* p0,
                                                const unsigned short* p1) {
    union { v4u q[2]; v16bf v; } u;
    u.q[0] = *(const v4u*)p0;
    u.q[1] = *(const v4u*)p1;
    return u.v;
}

// A fragment (16-bit A 16x32 layout, ISA 7.12.2): lane m = l&15, hi = l>>4,
// runs at K = hi*8 and K = 16 + hi*8 in a row-major LDS tile.
__device__ __forceinline__ v16bf ldsA(const unsigned short* base, int stride,
                                      int lane, int kbase) {
    const int m  = lane & 15;
    const int hi = lane >> 4;
    const unsigned short* p0 = base + m * stride + kbase + hi * 8;
    return load_frag_pair(p0, p0 + 16);
}

// B fragment from a staged tile (column-major [N][KT]): lane n = tile*16+(l&15),
// runs at K = hi*8 and K = 16 + hi*8 (matches A layout).
__device__ __forceinline__ v16bf ldsB(const unsigned short* wt, int ntile, int lane) {
    const int n  = ntile * 16 + (lane & 15);
    const int hi = lane >> 4;
    const unsigned short* p = wt + n * KT + hi * 8;
    return load_frag_pair(p, p + 16);
}

// stage one pre-packed 8 KB bf16 B tile: global -> LDS.
// Async CDNA5 DMA path (ASYNCcnt) with synchronous b128 fallback.
__device__ __forceinline__ void stage_tile(unsigned short* wtbuf,
                                           const unsigned short* __restrict__ Wimg,
                                           int ks, int tid) {
    const v4u* src = (const v4u*)(Wimg + (size_t)ks * TILE_HALVES);
    v4u*       dst = (v4u*)wtbuf;
#if USE_ASYNC
#pragma unroll
    for (int j = 0; j < (TILE_HALVES * 2 / 16) / NTHREADS; ++j) {   // 2 iters
        const int i = tid + j * NTHREADS;
        __builtin_amdgcn_global_load_async_to_lds_b128(
            (gvi4*)(src + i), (lvi4*)(dst + i), 0, 0);
    }
#else
#pragma unroll
    for (int j = 0; j < (TILE_HALVES * 2 / 16) / NTHREADS; ++j) {
        const int i = tid + j * NTHREADS;
        dst[i] = src[i];
    }
#endif
}

__device__ __forceinline__ void wait_async() {
#if USE_ASYNC
    __builtin_amdgcn_s_wait_asynccnt(0);
#endif
}

// [RT x K1] x [K1 x 128] GEMM. 8 waves each own a 16-col tile, 2 row-tiles of
// f32 accumulators per wave. Weight tiles double-buffered through async DMA.
template <int K1>
__device__ __forceinline__ void gemm_async(const unsigned short* aT, int astride,
                                           const unsigned short* __restrict__ Wimg,
                                           unsigned short* wt,   // 2 x TILE_HALVES
                                           int tid, int lane, int wv,
                                           v8f* acc, bool prestaged) {
    constexpr int NK = K1 / KT;
    if (!prestaged) stage_tile(wt, Wimg, 0, tid);
    for (int ks = 0; ks < NK; ++ks) {
        wait_async();                 // tile ks landed (per-wave ASYNCcnt)
        __syncthreads();              // ... and visible to every wave
        if (ks + 1 < NK)
            stage_tile(wt + ((ks + 1) & 1) * TILE_HALVES, Wimg, ks + 1, tid);
        const unsigned short* w = wt + (ks & 1) * TILE_HALVES;
        const v16bf b = ldsB(w, wv, lane);
#pragma unroll
        for (int rt = 0; rt < 2; ++rt) {
            const v16bf a = ldsA(aT + rt * 16 * astride, astride, lane, ks * KT);
            acc[rt] = __builtin_amdgcn_wmma_f32_16x16x32_bf16(
                false, a, false, b, (short)0, acc[rt], false, false);
        }
        __syncthreads();              // everyone done reading buf[ks&1]
    }
}

// ---------------------------------------------------------------------------
// One-time weight pack: fp32 [M][K1][128] -> bf16 B-fragment image:
// per layer, per k-tile ks: chunk[n*KT + kk] = bf16(W[(ks*KT+kk)*128 + n])
// ---------------------------------------------------------------------------
__global__ __launch_bounds__(NTHREADS)
void pack_weights(const float* __restrict__ W, unsigned short* __restrict__ out,
                  int K1) {
    const size_t total = (size_t)MM * K1 * DD;
    const size_t i = (size_t)blockIdx.x * NTHREADS + threadIdx.x;
    if (i >= total) return;
    const int per_layer = K1 * DD;
    const int l  = (int)(i / per_layer);
    const int r  = (int)(i % per_layer);
    const int ks = r / (KT * DD);
    const int t  = r % (KT * DD);
    const int kk = t / DD;
    const int n  = t % DD;       // consecutive i -> consecutive n: coalesced read
    out[(size_t)l * per_layer + (size_t)ks * TILE_HALVES + n * KT + kk] =
        f2bf(W[(size_t)l * per_layer + (size_t)(ks * KT + kk) * DD + n]);
}

// ---------------------------------------------------------------------------
// Edge model: e = LN(relu([x[row]|x[col]|ea] @ We1 + b1) @ We2 + b2)
//             ea_out = ea + e ; agg[col] += e (scatter)
// ---------------------------------------------------------------------------
__global__ __launch_bounds__(NTHREADS)
void edge_kernel(const float* __restrict__ x,   const float* __restrict__ ea,
                 const int*   __restrict__ rowi, const int* __restrict__ coli,
                 const unsigned short* __restrict__ W1img,
                 const float* __restrict__ be1,
                 const unsigned short* __restrict__ W2img,
                 const float* __restrict__ be2,
                 const float* __restrict__ g,   const float* __restrict__ bt,
                 float* __restrict__ ea_out,    float* __restrict__ agg)
{
    __shared__ __align__(16) unsigned short feats[RT * 3 * DD];     // 24 KB (f32 out alias)
    __shared__ __align__(16) unsigned short wt[2 * TILE_HALVES];    // 16 KB double buffer
    __shared__ __align__(16) unsigned short hid[RT * DD];           // 8 KB
    __shared__ int   sidx[2][RT];
    __shared__ float smu[RT], srs[RT];
    __shared__ float sg[DD], sbt[DD];
    __shared__ float sred[NTHREADS], qred[NTHREADS];
    float* outt = (float*)feats;                                    // 16 KB alias

    const int tid  = threadIdx.x;
    const int lane = tid & 31;
    const int wv   = tid >> 5;
    const long long e0 = (long long)blockIdx.x * RT;

    stage_tile(wt, W1img, 0, tid);           // overlap tile0 DMA with gather
    if (tid < RT) { sidx[0][tid] = rowi[e0 + tid]; sidx[1][tid] = coli[e0 + tid]; }
    if (tid < DD) { sg[tid] = g[tid]; sbt[tid] = bt[tid]; }
    __syncthreads();

    // gather feats = [x[row] | x[col] | ea] as bf16, two columns at a time
    for (int i = tid; i < RT * (3 * DD / 2); i += NTHREADS) {
        const int r  = i / (3 * DD / 2);
        const int c2 = (i % (3 * DD / 2)) * 2;
        float2 v;
        if (c2 < DD)            v = *(const float2*)&x[(size_t)sidx[0][r] * DD + c2];
        else if (c2 < 2 * DD)   v = *(const float2*)&x[(size_t)sidx[1][r] * DD + (c2 - DD)];
        else                    v = *(const float2*)&ea[(e0 + r) * DD + (c2 - 2 * DD)];
        ((unsigned*)feats)[r * (3 * DD / 2) + (c2 >> 1)] = pack2bf(v.x, v.y);
    }
    __builtin_prefetch(W2img, 0, 1);         // global_prefetch_b8 toward GEMM2 weights

    v8f zero = {};
    v8f acc[2] = {zero, zero};
    gemm_async<3 * DD>(feats, 3 * DD, W1img, wt, tid, lane, wv, acc, true);

    // bias + relu -> bf16 hidden
    {
        const int n  = wv * 16 + (lane & 15);
        const int hi = lane >> 4;
        const float b1 = be1[n];
#pragma unroll
        for (int rt = 0; rt < 2; ++rt)
#pragma unroll
            for (int i2 = 0; i2 < 8; ++i2) {
                const float v = acc[rt][i2] + b1;
                hid[(rt * 16 + hi * 8 + i2) * DD + n] = f2bf(v > 0.f ? v : 0.f);
            }
    }
    __syncthreads();

    v8f acc2[2] = {zero, zero};
    gemm_async<DD>(hid, DD, W2img, wt, tid, lane, wv, acc2, false);

    // bias + write f32 tile (feats region is dead, alias it)
    {
        const int n  = wv * 16 + (lane & 15);
        const int hi = lane >> 4;
        const float b2 = be2[n];
#pragma unroll
        for (int rt = 0; rt < 2; ++rt)
#pragma unroll
            for (int i2 = 0; i2 < 8; ++i2)
                outt[(rt * 16 + hi * 8 + i2) * DD + n] = acc2[rt][i2] + b2;
    }
    __syncthreads();

    // LayerNorm statistics: 8 threads per row, then tree-combine
    {
        const int r = tid >> 3, sub = tid & 7;
        float s = 0.f, ss = 0.f;
#pragma unroll
        for (int d = sub * 16; d < sub * 16 + 16; ++d) {
            const float v = outt[r * DD + d];
            s += v; ss += v * v;
        }
        sred[tid] = s; qred[tid] = ss;
    }
    __syncthreads();
    if (tid < RT) {
        float S = 0.f, Q = 0.f;
#pragma unroll
        for (int j = 0; j < 8; ++j) { S += sred[tid * 8 + j]; Q += qred[tid * 8 + j]; }
        const float mu  = S * (1.f / DD);
        const float var = Q * (1.f / DD) - mu * mu;
        smu[tid] = mu;
        srs[tid] = rsqrtf(var + LN_EPS);
    }
    __syncthreads();

    // normalize, residual edge update, scatter-sum into agg
    for (int i = tid; i < RT * DD; i += NTHREADS) {
        const int r = i >> 7, d = i & 127;
        const float y = (outt[i] - smu[r]) * srs[r] * sg[d] + sbt[d];
        ea_out[(e0 + r) * DD + d] = ea[(e0 + r) * DD + d] + y;
        atomicAdd(&agg[(size_t)sidx[1][r] * DD + d], y);
    }
}

// ---------------------------------------------------------------------------
// Node model: n = LN(relu([x|agg] @ Wn1 + b1) @ Wn2 + b2) ; x_out = x + n
// ---------------------------------------------------------------------------
__global__ __launch_bounds__(NTHREADS)
void node_kernel(const float* __restrict__ x,   const float* __restrict__ agg,
                 const unsigned short* __restrict__ W1img,
                 const float* __restrict__ bn1,
                 const unsigned short* __restrict__ W2img,
                 const float* __restrict__ bn2,
                 const float* __restrict__ g,   const float* __restrict__ bt,
                 float* __restrict__ x_out)
{
    __shared__ __align__(16) unsigned short feats[RT * 2 * DD];     // 16 KB (f32 out alias)
    __shared__ __align__(16) unsigned short wt[2 * TILE_HALVES];    // 16 KB
    __shared__ __align__(16) unsigned short hid[RT * DD];           // 8 KB
    __shared__ float smu[RT], srs[RT];
    __shared__ float sg[DD], sbt[DD];
    __shared__ float sred[NTHREADS], qred[NTHREADS];
    float* outt = (float*)feats;

    const int tid  = threadIdx.x;
    const int lane = tid & 31;
    const int wv   = tid >> 5;
    const long long n0 = (long long)blockIdx.x * RT;

    stage_tile(wt, W1img, 0, tid);
    if (tid < DD) { sg[tid] = g[tid]; sbt[tid] = bt[tid]; }

    // feats = [x | agg] as bf16, two columns at a time
    for (int i = tid; i < RT * DD; i += NTHREADS) {   // RT*(2*DD/2)
        const int r  = i / DD;
        const int c2 = (i % DD) * 2;
        const float2 v = (c2 < DD)
            ? *(const float2*)&x[(n0 + r) * DD + c2]
            : *(const float2*)&agg[(n0 + r) * DD + (c2 - DD)];
        ((unsigned*)feats)[r * DD + (c2 >> 1)] = pack2bf(v.x, v.y);
    }
    __builtin_prefetch(W2img, 0, 1);

    v8f zero = {};
    v8f acc[2] = {zero, zero};
    gemm_async<2 * DD>(feats, 2 * DD, W1img, wt, tid, lane, wv, acc, true);

    {
        const int n  = wv * 16 + (lane & 15);
        const int hi = lane >> 4;
        const float b1 = bn1[n];
#pragma unroll
        for (int rt = 0; rt < 2; ++rt)
#pragma unroll
            for (int i2 = 0; i2 < 8; ++i2) {
                const float v = acc[rt][i2] + b1;
                hid[(rt * 16 + hi * 8 + i2) * DD + n] = f2bf(v > 0.f ? v : 0.f);
            }
    }
    __syncthreads();

    v8f acc2[2] = {zero, zero};
    gemm_async<DD>(hid, DD, W2img, wt, tid, lane, wv, acc2, false);

    {
        const int n  = wv * 16 + (lane & 15);
        const int hi = lane >> 4;
        const float b2 = bn2[n];
#pragma unroll
        for (int rt = 0; rt < 2; ++rt)
#pragma unroll
            for (int i2 = 0; i2 < 8; ++i2)
                outt[(rt * 16 + hi * 8 + i2) * DD + n] = acc2[rt][i2] + b2;
    }
    __syncthreads();

    {
        const int r = tid >> 3, sub = tid & 7;
        float s = 0.f, ss = 0.f;
#pragma unroll
        for (int d = sub * 16; d < sub * 16 + 16; ++d) {
            const float v = outt[r * DD + d];
            s += v; ss += v * v;
        }
        sred[tid] = s; qred[tid] = ss;
    }
    __syncthreads();
    if (tid < RT) {
        float S = 0.f, Q = 0.f;
#pragma unroll
        for (int j = 0; j < 8; ++j) { S += sred[tid * 8 + j]; Q += qred[tid * 8 + j]; }
        const float mu  = S * (1.f / DD);
        const float var = Q * (1.f / DD) - mu * mu;
        smu[tid] = mu;
        srs[tid] = rsqrtf(var + LN_EPS);
    }
    __syncthreads();

    for (int i = tid; i < RT * DD; i += NTHREADS) {
        const int r = i >> 7, d = i & 127;
        const float y = (outt[i] - smu[r]) * srs[r] * sg[d] + sbt[d];
        x_out[(n0 + r) * DD + d] = x[(n0 + r) * DD + d] + y;
    }
}

// ---------------------------------------------------------------------------
// Host-side orchestration (graph-capture safe: kernels + memset/memcpy async)
// ---------------------------------------------------------------------------
extern "C" void kernel_launch(void* const* d_in, const int* in_sizes, int n_in,
                              void* d_out, int out_size, void* d_ws, size_t ws_size,
                              hipStream_t stream) {
    (void)in_sizes; (void)n_in; (void)out_size; (void)ws_size;

    const float* x_in  = (const float*)d_in[0];
    const float* ea_in = (const float*)d_in[1];
    const int*   ei    = (const int*)d_in[2];
    const float* We1   = (const float*)d_in[3];
    const float* be1   = (const float*)d_in[4];
    const float* We2   = (const float*)d_in[5];
    const float* be2   = (const float*)d_in[6];
    const float* ge    = (const float*)d_in[7];
    const float* bte   = (const float*)d_in[8];
    const float* Wn1   = (const float*)d_in[9];
    const float* bn1   = (const float*)d_in[10];
    const float* Wn2   = (const float*)d_in[11];
    const float* bn2   = (const float*)d_in[12];
    const float* gn    = (const float*)d_in[13];
    const float* btn   = (const float*)d_in[14];

    const int* rowi = ei;        // edge_index[0] (source)
    const int* coli = ei + EE;   // edge_index[1] (destination)

    // workspace: agg | x ping | x pong | ea ping | ea pong | bf16 weight images
    float* agg = (float*)d_ws;
    float* xb0 = agg + (size_t)NN * DD;
    float* xb1 = xb0 + (size_t)NN * DD;
    float* eb0 = xb1 + (size_t)NN * DD;
    float* eb1 = eb0 + (size_t)EE * DD;
    unsigned short* we1p = (unsigned short*)(eb1 + (size_t)EE * DD); // 16B aligned
    unsigned short* we2p = we1p + (size_t)MM * 3 * DD * DD;
    unsigned short* wn1p = we2p + (size_t)MM * DD * DD;
    unsigned short* wn2p = wn1p + (size_t)MM * 2 * DD * DD;

    // one-time weight packing fp32 -> bf16 B-fragment images
    {
        const int t1 = MM * 3 * DD * DD, t2 = MM * DD * DD, t3 = MM * 2 * DD * DD;
        pack_weights<<<(t1 + NTHREADS - 1) / NTHREADS, NTHREADS, 0, stream>>>(We1, we1p, 3 * DD);
        pack_weights<<<(t2 + NTHREADS - 1) / NTHREADS, NTHREADS, 0, stream>>>(We2, we2p, DD);
        pack_weights<<<(t3 + NTHREADS - 1) / NTHREADS, NTHREADS, 0, stream>>>(Wn1, wn1p, 2 * DD);
        pack_weights<<<(t2 + NTHREADS - 1) / NTHREADS, NTHREADS, 0, stream>>>(Wn2, wn2p, DD);
    }

    const float* xc = x_in;
    const float* ec = ea_in;

    for (int l = 0; l < MM; ++l) {
        float* xn = (l & 1) ? xb1 : xb0;
        float* en = (l & 1) ? eb1 : eb0;

        (void)hipMemsetAsync(agg, 0, (size_t)NN * DD * sizeof(float), stream);

        edge_kernel<<<EE / RT, NTHREADS, 0, stream>>>(
            xc, ec, rowi, coli,
            we1p + (size_t)l * 3 * DD * DD, be1 + (size_t)l * DD,
            we2p + (size_t)l * DD * DD,     be2 + (size_t)l * DD,
            ge + (size_t)l * DD,            bte + (size_t)l * DD,
            en, agg);

        node_kernel<<<NN / RT, NTHREADS, 0, stream>>>(
            xc, agg,
            wn1p + (size_t)l * 2 * DD * DD, bn1 + (size_t)l * DD,
            wn2p + (size_t)l * DD * DD,     bn2 + (size_t)l * DD,
            gn + (size_t)l * DD,            btn + (size_t)l * DD,
            xn);

        xc = xn;
        ec = en;
    }

    // d_out = [x (N*D) | edge_attr (E*D)], fp32
    (void)hipMemcpyAsync(d_out, xc, (size_t)NN * DD * sizeof(float),
                         hipMemcpyDeviceToDevice, stream);
    (void)hipMemcpyAsync((float*)d_out + (size_t)NN * DD, ec,
                         (size_t)EE * DD * sizeof(float),
                         hipMemcpyDeviceToDevice, stream);
}
